// RNN_CrossAttention_4458176053702
// MI455X (gfx1250) — compile-verified, matrix operand order checked by reading, not run on hardware
//
#include <hip/hip_runtime.h>
#include <cstddef>

// ---------------------------------------------------------------------------
// Problem constants (match reference)
// ---------------------------------------------------------------------------
static constexpr int B_  = 16;
static constexpr int T_  = 128;
static constexpr int D_  = 128;   // data dim
static constexpr int F_  = 128;   // state "rows"
static constexpr int ETA = 64;    // nonlinear expansion dim
static constexpr int NS  = 128;   // state cols / attention feature dim

#define DEV __device__ __forceinline__

typedef __bf16 bf16;
typedef bf16  v16bf __attribute__((ext_vector_type(16)));
typedef bf16  v8bf  __attribute__((ext_vector_type(8)));
typedef bf16  v2bf  __attribute__((ext_vector_type(2)));
typedef float v8f   __attribute__((ext_vector_type(8)));

#if __has_builtin(__builtin_amdgcn_tanhf)
#define TANHF(v) __builtin_amdgcn_tanhf(v)
#else
#define TANHF(v) tanhf(v)
#endif

// Opaque a 32-bit LDS *offset* (not the pointer!): defeats cross-iteration
// CSE/hoisting of fragment loads while keeping the base object known to the
// compiler, so addressing stays in the LDS address space (ds_load, not flat).
DEV int opq(int v) { asm volatile("" : "+v"(v)); return v; }

// ---------------------------------------------------------------------------
// WMMA helpers (CDNA5, wave32, V_WMMA_F32_16X16X32_BF16)
// ---------------------------------------------------------------------------

// Load one 16x32 bf16 operand fragment from row-major [M][K] LDS.
// Per ISA layout: lane-half 0 holds K = {0..7, 16..23}+k0, half 1 holds
// {8..15, 24..31}+k0; row = rowBase + (lane&15). The same loader serves
// B-operands when the matrix is stored transposed ([N][K]).
DEV v16bf load_frag(const bf16* base, int ld, int rowBase, int k0, int lane) {
    const int l  = lane & 15;
    const int hi = lane >> 4;
    const bf16* p = base + (size_t)(rowBase + l) * ld + k0 + hi * 8;
    v8bf a = *(const v8bf*)(p);
    v8bf b = *(const v8bf*)(p + 16);
    return __builtin_shufflevector(a, b, 0, 1, 2, 3, 4, 5, 6, 7,
                                         8, 9, 10, 11, 12, 13, 14, 15);
}

DEV v8f wmma_bf16(v16bf a, v16bf b, v8f c) {
    // (neg_a, A, neg_b, B, c_mod, C, reuse_a, reuse_b)
    return __builtin_amdgcn_wmma_f32_16x16x32_bf16(false, a, false, b,
                                                   (short)0, c, false, false);
}

template <int K>
DEV void load_afrags(const bf16* A, int lda, int mb, int lane, v16bf af[K / 32]) {
#pragma unroll
    for (int i = 0; i < K / 32; ++i)
        af[i] = load_frag(A, lda, mb * 16, i * 32, lane);
}

// Accumulate all 8 column tiles of one 16x128 row block (keeps independent
// WMMA chains in flight -> hazard windows filled by neighboring tiles).
template <int NK>
DEV void mm_tiles(const v16bf af[NK], const bf16* BT, int ldb, int lane,
                  v8f acc[8]) {
#pragma unroll
    for (int n = 0; n < 8; ++n) {
        v8f c = {};
#pragma unroll
        for (int i = 0; i < NK; ++i)
            c = wmma_bf16(af[i], load_frag(BT, ldb, n * 16, i * 32, lane), c);
        acc[n] = c;
    }
}

// Store a 16x128 row block of f32 accumulators as bf16.
// TRANS=false: row-major (b16 scatter, row stride). TRANS=true: per-lane
// contiguous 8 elements -> single ds_store_b128 per tile.
template <bool TRANS>
DEV void store_tiles(bf16* C, int ldc, int mb, int lane, const v8f acc[8]) {
    const int l  = lane & 15;
    const int hi = lane >> 4;
#pragma unroll
    for (int n = 0; n < 8; ++n) {
        if (TRANS) {
            v8bf pk;
#pragma unroll
            for (int r = 0; r < 8; ++r) pk[r] = (bf16)acc[n][r];
            *(v8bf*)&C[(size_t)(n * 16 + l) * ldc + mb * 16 + hi * 8] = pk;
        } else {
#pragma unroll
            for (int r = 0; r < 8; ++r)
                C[(size_t)(mb * 16 + r + hi * 8) * ldc + n * 16 + l] = (bf16)acc[n][r];
        }
    }
}

template <int K, bool TRANS>
DEV void mm_rowblock_store(const bf16* A, int lda, const bf16* BT, int ldb,
                           bf16* C, int ldc, int mb, int lane) {
    v16bf af[K / 32];
    load_afrags<K>(A, lda, mb, lane, af);
    v8f acc[8];
    mm_tiles<K / 32>(af, BT, ldb, lane, acc);
    store_tiles<TRANS>(C, ldc, mb, lane, acc);
}

// Fused: C1 = A x B1 (row-major store), C2 = A x B2 (transposed store);
// A fragments loaded once for both products.
template <int K>
DEV void mm_rowblock_store2(const bf16* A, int lda,
                            const bf16* BT1, const bf16* BT2, int ldb,
                            bf16* C1, bf16* C2, int ldc, int mb, int lane) {
    v16bf af[K / 32];
    load_afrags<K>(A, lda, mb, lane, af);
    v8f acc[8];
    mm_tiles<K / 32>(af, BT1, ldb, lane, acc);
    store_tiles<false>(C1, ldc, mb, lane, acc);
    mm_tiles<K / 32>(af, BT2, ldb, lane, acc);
    store_tiles<true>(C2, ldc, mb, lane, acc);
}

// Row softmax over a 16x128 accumulator row-block.
// C-layout: for accumulator register r, a matrix row's 128 entries live in
// acc[0..7][r] across the 16 lanes of the lane-half -> reduce over n, then
// shfl_xor over offsets {1,2,4,8} (stays within each aligned 16-lane group).
DEV void softmax_rows(v8f acc[8], float scale) {
#pragma unroll
    for (int r = 0; r < 8; ++r) {
        float m = -3.4e38f;
#pragma unroll
        for (int n = 0; n < 8; ++n) {
            acc[n][r] *= scale;
            m = fmaxf(m, acc[n][r]);
        }
#pragma unroll
        for (int off = 1; off < 16; off <<= 1)
            m = fmaxf(m, __shfl_xor(m, off, 32));
        float s = 0.0f;
#pragma unroll
        for (int n = 0; n < 8; ++n) {
            float e = __expf(acc[n][r] - m);
            acc[n][r] = e;
            s += e;
        }
#pragma unroll
        for (int off = 1; off < 16; off <<= 1)
            s += __shfl_xor(s, off, 32);
        const float inv = 1.0f / s;
#pragma unroll
        for (int n = 0; n < 8; ++n) acc[n][r] *= inv;
    }
}

// ---------------------------------------------------------------------------
// Weight prep: fp32 [K][N] -> bf16 transposed [N][K] in workspace
// ---------------------------------------------------------------------------
__global__ void transpose_to_bf16(const float* __restrict__ src,
                                  bf16* __restrict__ dst, int K, int N) {
    int idx = blockIdx.x * blockDim.x + threadIdx.x;
    if (idx < K * N) {
        int k = idx / N, n = idx % N;
        dst[n * K + k] = (bf16)src[idx];
    }
}

// ---------------------------------------------------------------------------
// LDS layout (element offsets into one carved array; single object => the
// compiler cannot prove the weight region is unmodified by in-loop stores,
// so fragment loads are never hoisted out of the t-loop).
// ---------------------------------------------------------------------------
static constexpr int OFF_WQO = 0;                      // [128][64]
static constexpr int OFF_WVO = OFF_WQO + 128 * 64;     // [128][64]
static constexpr int OFF_WKS = OFF_WVO + 128 * 64;     // [128][64]
static constexpr int OFF_WKO = OFF_WKS + 128 * 64;     // [128][128]
static constexpr int OFF_WQS = OFF_WKO + 128 * 128;    // [128][128]
static constexpr int OFF_WVS = OFF_WQS + 128 * 128;    // [128][128]
static constexpr int W_ELEMS = OFF_WVS + 128 * 128;    // 73728
static constexpr int OFF_S   = W_ELEMS;                // state [128][128]
static constexpr int OFF_U   = OFF_S + 128 * 128;      // u [128][64]
static constexpr int OFF_Q   = OFF_U + 128 * 64;       // q / w / qs / ws
static constexpr int OFF_V   = OFF_Q + 128 * 128;      // v^T / vs^T
static constexpr int OFF_K   = OFF_V + 128 * 128;      // k / ks
static constexpr int LDS_ELEMS = OFF_K + 128 * 128;    // 147456 (~288 KB)

// ---------------------------------------------------------------------------
// Main kernel: one workgroup per batch, 8 waves, T iterated internally.
// Weights (bf16, transposed) + state live in LDS for the whole sequence.
// ---------------------------------------------------------------------------
__global__ __launch_bounds__(256)
void rnn_xattn_kernel(const float* __restrict__ x, const float* __restrict__ h0,
                      const float* __restrict__ nl,
                      const bf16* __restrict__ wsW,  // packed transposed weights
                      float* __restrict__ out) {
    __shared__ __align__(16) bf16 smem[LDS_ELEMS];
    __shared__ float sNl[ETA];

    bf16* sS   = smem + OFF_S;
    bf16* sU   = smem + OFF_U;
    bf16* bufQ = smem + OFF_Q;
    bf16* bufV = smem + OFF_V;
    bf16* bufK = smem + OFF_K;

    const int b    = blockIdx.x;
    const int tid  = threadIdx.x;
    const int wave = tid >> 5;
    const int lane = tid & 31;

    // --- init: weights -> LDS (contiguous b128 copy), h0 -> LDS, nl -> LDS
    {
        const v8bf* src = (const v8bf*)wsW;
        v8bf* dst = (v8bf*)smem;
        for (int i = tid; i < W_ELEMS / 8; i += 256) dst[i] = src[i];
    }
    for (int i = tid; i < F_ * NS; i += 256)
        sS[i] = (bf16)h0[(size_t)b * (F_ * NS) + i];
    if (tid < ETA) sNl[tid] = nl[tid];
    __syncthreads();

    for (int t = 0; t < T_; ++t) {
        // Opaque *offsets* (stay in LDS address space; no cross-iteration CSE)
        const bf16* wqoT = smem + opq(OFF_WQO);
        const bf16* wvoT = smem + opq(OFF_WVO);
        const bf16* wksT = smem + opq(OFF_WKS);
        const bf16* wkoT = smem + opq(OFF_WKO);
        const bf16* wqsT = smem + opq(OFF_WQS);
        const bf16* wvsT = smem + opq(OFF_WVS);

        // ---- phase 1: u = tanh(x_t[d] * nl[e])  (2 threads per row) ----
        {
            const int row = tid >> 1;
            const int c0  = (tid & 1) * 32;
            const float xv = x[((size_t)b * T_ + t) * D_ + row];
            v2bf* du = (v2bf*)&sU[row * ETA + c0];
#pragma unroll
            for (int c = 0; c < 32; c += 2) {
                v2bf pk;
                pk[0] = (bf16)TANHF(xv * sNl[c0 + c]);
                pk[1] = (bf16)TANHF(xv * sNl[c0 + c + 1]);
                du[c >> 1] = pk;
            }
        }
        __syncthreads();  // (A) sU ready; all waves past last step's phase 6

        // ---- phase 2: q = u*Wqo, v^T = (u*Wvo)^T (fused A), k = s*Wko ----
        mm_rowblock_store2<64>(sU, ETA, wqoT, wvoT, ETA, bufQ, bufV, NS, wave, lane);
        mm_rowblock_store<128, false>(sS, NS, wkoT, NS, bufK, NS, wave, lane);
        __syncthreads();  // (B)

        // ---- phase 3: g = q*k^T / NS ; softmax rows ; w -> bufQ (own rows) ----
        v8f acc[8];
        {
            v16bf af[4];
            load_afrags<128>(bufQ, NS, wave, lane, af);
            mm_tiles<4>(af, bufK, NS, lane, acc);
        }
        softmax_rows(acc, 1.0f / (float)NS);
        store_tiles<false>(bufQ, NS, wave, lane, acc);  // own row-block only

        // ---- phase 4: out = w * v  -> global fp32 (coalesced) ----
        {
            v16bf af[4];
            load_afrags<128>(bufQ, NS, wave, lane, af);
            mm_tiles<4>(af, bufV, NS, lane, acc);
            const int l  = lane & 15;
            const int hi = lane >> 4;
            float* op = out + (((size_t)b * T_ + t) * (size_t)D_) * NS;
#pragma unroll
            for (int n = 0; n < 8; ++n)
#pragma unroll
                for (int r = 0; r < 8; ++r)
                    op[(size_t)(wave * 16 + r + hi * 8) * NS + n * 16 + l] = acc[n][r];
        }
        __syncthreads();  // (C) q/v/k dead; safe to overwrite buffers

        // ---- phase 5: qs = s*Wqs, vs^T = (s*Wvs)^T (fused A), ks = u*Wks ----
        mm_rowblock_store2<128>(sS, NS, wqsT, wvsT, NS, bufQ, bufV, NS, wave, lane);
        mm_rowblock_store<64, false>(sU, ETA, wksT, ETA, bufK, NS, wave, lane);
        __syncthreads();  // (D) all sS reads of this step are done

        // ---- phase 6: gs = qs*ks^T / NS ; softmax ; h_new = ws*vs -> sS ----
        {
            v16bf af[4];
            load_afrags<128>(bufQ, NS, wave, lane, af);
            mm_tiles<4>(af, bufK, NS, lane, acc);
        }
        softmax_rows(acc, 1.0f / (float)NS);
        store_tiles<false>(bufQ, NS, wave, lane, acc);                      // ws
        mm_rowblock_store<128, false>(bufQ, NS, bufV, NS, sS, NS, wave, lane); // h_new
        __syncthreads();  // (E) next step reads full sS
    }
}

// ---------------------------------------------------------------------------
// Launch
// ---------------------------------------------------------------------------
extern "C" void kernel_launch(void* const* d_in, const int* in_sizes, int n_in,
                              void* d_out, int out_size, void* d_ws, size_t ws_size,
                              hipStream_t stream) {
    (void)in_sizes; (void)n_in; (void)out_size; (void)ws_size;

    const float* x   = (const float*)d_in[0];
    const float* h0  = (const float*)d_in[1];
    const float* Wqo = (const float*)d_in[2];  // [ETA][NS]
    const float* Wvo = (const float*)d_in[3];  // [ETA][NS]
    const float* Wko = (const float*)d_in[4];  // [NS][NS]
    const float* Wqs = (const float*)d_in[5];  // [NS][NS]
    const float* Wvs = (const float*)d_in[6];  // [NS][NS]
    const float* Wks = (const float*)d_in[7];  // [ETA][NS]
    const float* nl  = (const float*)d_in[8];  // [1][ETA]

    // bf16 transposed weights, packed contiguously in workspace in the same
    // order the kernel carves its LDS weight block.
    bf16* ws = (bf16*)d_ws;
    const int thr = 256;
    transpose_to_bf16<<<(ETA * NS + thr - 1) / thr, thr, 0, stream>>>(Wqo, ws + OFF_WQO, ETA, NS);
    transpose_to_bf16<<<(ETA * NS + thr - 1) / thr, thr, 0, stream>>>(Wvo, ws + OFF_WVO, ETA, NS);
    transpose_to_bf16<<<(ETA * NS + thr - 1) / thr, thr, 0, stream>>>(Wks, ws + OFF_WKS, ETA, NS);
    transpose_to_bf16<<<(NS * NS + thr - 1) / thr, thr, 0, stream>>>(Wko, ws + OFF_WKO, NS, NS);
    transpose_to_bf16<<<(NS * NS + thr - 1) / thr, thr, 0, stream>>>(Wqs, ws + OFF_WQS, NS, NS);
    transpose_to_bf16<<<(NS * NS + thr - 1) / thr, thr, 0, stream>>>(Wvs, ws + OFF_WVS, NS, NS);

    rnn_xattn_kernel<<<B_, 256, 0, stream>>>(x, h0, nl, ws, (float*)d_out);
}
